// PreNormedHGTCavAttention_70927089926445
// MI455X (gfx1250) — matrix operand order repaired and checked by reading, not compiled, not run on hardware
//
#include <hip/hip_runtime.h>
#include <hip/hip_bf16.h>

// ---------------------------------------------------------------------------
// PreNormedHGTCavAttention for MI455X (gfx1250, wave32, WMMA bf16)
//
// Layouts (all workspace tensors):
//   xn   : (b*l, hw, 256) bf16      LayerNormed input, (b,l)-major
//   Q/K/V: (b*l, hw, 256) bf16      inner = m*32 + dh
//   kt/vt: (b, t_i, m, j, hw, 32) bf16
//   ao   : (b*l, hw, 256) bf16      attention output before final projection
// Output d_out is (B,L,H,W,DIM) f32 == (b*l, hw, 256) flat -> no transpose.
// ---------------------------------------------------------------------------

typedef __attribute__((ext_vector_type(16))) __bf16 bf16x16;
typedef __attribute__((ext_vector_type(8)))  __bf16 bf16x8;
typedef __attribute__((ext_vector_type(8)))  float  f32x8;

#define DEV static __device__ __forceinline__

// A-fragment (16x32 bf16): lane<16 holds K {0..15}, lane>=16 {16..31}.
// Caller passes p already shifted by (lane<16 ? 0 : 16); chunks at p and p+8.
DEV bf16x16 load_frag_gap16(const __bf16* p) {
  bf16x8 lo = *(const bf16x8*)p;
  bf16x8 hi = *(const bf16x8*)(p + 8);
  bf16x16 r;
#pragma unroll
  for (int i = 0; i < 8; ++i) { r[i] = lo[i]; r[i + 8] = hi[i]; }
  return r;
}

// B-fragment (32x16 bf16): lane holds column n = lane%16, 16 consecutive K.
// Caller passes p already shifted by (lane<16 ? 0 : 16); 16 contiguous bf16.
DEV bf16x16 load_frag_contig(const __bf16* p) {
  bf16x8 lo = *(const bf16x8*)p;
  bf16x8 hi = *(const bf16x8*)(p + 8);
  bf16x16 r;
#pragma unroll
  for (int i = 0; i < 8; ++i) { r[i] = lo[i]; r[i + 8] = hi[i]; }
  return r;
}

DEV f32x8 wmma_bf16(bf16x16 a, bf16x16 b, f32x8 c) {
  return __builtin_amdgcn_wmma_f32_16x16x32_bf16(
      /*neg_a=*/false, a, /*neg_b=*/false, b,
      /*c_mod=*/(short)0, c, /*reuse_a=*/false, /*reuse_b=*/false);
}

// ---------------- small converters --------------------------------------------

__global__ void __launch_bounds__(256) k_cvt(const float* __restrict__ src,
                                             __bf16* __restrict__ dst, int n) {
  int i = blockIdx.x * 256 + threadIdx.x;
  if (i < n) dst[i] = (__bf16)src[i];
}

// relation_msg (4,8,32,32): transpose last two dims so B-columns are contiguous
__global__ void __launch_bounds__(1024) k_trel(const float* __restrict__ src,
                                               __bf16* __restrict__ dst) {
  int rm = blockIdx.x;                 // 0..31  (rel*8 + head)
  int p = threadIdx.x >> 5;            // 0..31
  int c = threadIdx.x & 31;            // 0..31
  dst[(size_t)rm * 1024 + c * 32 + p] = (__bf16)src[(size_t)rm * 1024 + p * 32 + c];
}

// ---------------- LayerNorm (wave per token) ----------------------------------

__global__ void __launch_bounds__(256) k_ln(const float* __restrict__ x,
                                            const float* __restrict__ gw,
                                            const float* __restrict__ gb,
                                            __bf16* __restrict__ xn) {
  const int lane = threadIdx.x & 31;
  const size_t tok = (size_t)blockIdx.x * 8 + (threadIdx.x >> 5);
  const float* p = x + tok * 256;
  float v[8];
  float s = 0.f;
#pragma unroll
  for (int e = 0; e < 8; ++e) { v[e] = p[e * 32 + lane]; s += v[e]; }
#pragma unroll
  for (int o = 16; o > 0; o >>= 1) s += __shfl_xor(s, o, 32);
  const float mu = s * (1.0f / 256.0f);
  float s2 = 0.f;
#pragma unroll
  for (int e = 0; e < 8; ++e) { float d = v[e] - mu; s2 += d * d; }
#pragma unroll
  for (int o = 16; o > 0; o >>= 1) s2 += __shfl_xor(s2, o, 32);
  const float rstd = rsqrtf(s2 * (1.0f / 256.0f) + 1e-5f);
  __bf16* q = xn + tok * 256;
#pragma unroll
  for (int e = 0; e < 8; ++e) {
    const int c = e * 32 + lane;
    q[c] = (__bf16)((v[e] - mu) * rstd * gw[c] + gb[c]);
  }
}

// ---------------- fused Q/K/V projection GEMM ---------------------------------
// grid (128, 20): per (b,l), 64 M-tiles x 16 N-tiles; wave = one 16x16 tile,
// computes Q,K,V together so the A fragment is loaded once (24 wmma / wave).

__global__ void __launch_bounds__(256) k_qkv(
    const __bf16* __restrict__ xn,
    const __bf16* __restrict__ wq, const __bf16* __restrict__ wk,
    const __bf16* __restrict__ wv,
    const float* __restrict__ qb, const float* __restrict__ kb,
    const float* __restrict__ vb,
    const int* __restrict__ pe,
    __bf16* __restrict__ Qo, __bf16* __restrict__ Ko, __bf16* __restrict__ Vo) {
  const int lane = threadIdx.x & 31;
  const int tile = blockIdx.x * 8 + (threadIdx.x >> 5);   // 0..1023
  const int mt = tile >> 4, nt = tile & 15;
  const int bl = blockIdx.y;                              // 0..19
  const int t = pe[bl * 3072 + 2];                        // node type
  const int arow = mt * 16 + (lane & 15);
  const int ncol = nt * 16 + (lane & 15);
  const __bf16* ap = xn + ((size_t)bl * 1024 + arow) * 256 + ((lane < 16) ? 0 : 16);
  const size_t boff = (size_t)t * 65536 + (size_t)ncol * 256 + ((lane < 16) ? 0 : 16);
  const __bf16* bq = wq + boff;
  const __bf16* bk = wk + boff;
  const __bf16* bv = wv + boff;
  f32x8 cq = {}, ck = {}, cv = {};
#pragma unroll
  for (int k0 = 0; k0 < 256; k0 += 32) {
    bf16x16 a = load_frag_gap16(ap + k0);
    cq = wmma_bf16(a, load_frag_contig(bq + k0), cq);
    ck = wmma_bf16(a, load_frag_contig(bk + k0), ck);
    cv = wmma_bf16(a, load_frag_contig(bv + k0), cv);
  }
  const float biq = qb[t * 256 + ncol];
  const float bik = kb[t * 256 + ncol];
  const float biv = vb[t * 256 + ncol];
#pragma unroll
  for (int r = 0; r < 8; ++r) {
    const int mrow = mt * 16 + ((lane < 16) ? r : r + 8);
    const size_t o = ((size_t)bl * 1024 + mrow) * 256 + ncol;
    Qo[o] = (__bf16)(cq[r] + biq);
    Ko[o] = (__bf16)(ck[r] + bik);
    Vo[o] = (__bf16)(cv[r] + biv);
  }
}

// ---------------- relation transforms: kt = W_att . k_j ; vt = W_msg^T . v_j --
// grid (16, 320, 2): y = (b,m,j,t_i); M=1024 over hw, N=32, K=32 (1 wmma/wave).

__global__ void __launch_bounds__(256) k_rel(
    const __bf16* __restrict__ Kin, const __bf16* __restrict__ Vin,
    const __bf16* __restrict__ ra, const __bf16* __restrict__ rmT,
    const int* __restrict__ pe,
    __bf16* __restrict__ kt, __bf16* __restrict__ vt) {
  const int lane = threadIdx.x & 31;
  const int tile = blockIdx.x * 8 + (threadIdx.x >> 5);   // 0..127
  const int mt = tile >> 1, nt = tile & 1;
  const int y = blockIdx.y;
  const int ti = y & 1;
  const int j = (y >> 1) % 5;
  const int m = (y / 10) & 7;
  const int b = y / 80;
  const int tj = pe[(b * 5 + j) * 3072 + 2];
  const int rel = ti * 2 + tj;
  const int isV = blockIdx.z;
  const __bf16* src = isV ? Vin : Kin;
  const __bf16* W = (isV ? rmT : ra) + (size_t)(rel * 8 + m) * 1024;
  const int hwrow = mt * 16 + (lane & 15);
  bf16x16 a = load_frag_gap16(src + ((size_t)(b * 5 + j) * 1024 + hwrow) * 256 +
                              m * 32 + ((lane < 16) ? 0 : 16));
  const int ncol = nt * 16 + (lane & 15);
  bf16x16 bfr = load_frag_contig(W + ncol * 32 + ((lane < 16) ? 0 : 16));
  f32x8 c = {};
  c = wmma_bf16(a, bfr, c);
  __bf16* dst = (isV ? vt : kt) + (size_t)(((b * 2 + ti) * 8 + m) * 5 + j) * 32768;
#pragma unroll
  for (int e = 0; e < 8; ++e) {
    const int hw = mt * 16 + ((lane < 16) ? e : e + 8);
    dst[(size_t)hw * 32 + ncol] = (__bf16)c[e];
  }
}

// ---------------- attention: dot + softmax(L=5) + weighted sum ---------------
// wave = one (b, hw, m, i); lane = dh channel.

__global__ void __launch_bounds__(256) k_attn(
    const __bf16* __restrict__ Qb, const __bf16* __restrict__ kt,
    const __bf16* __restrict__ vt, const int* __restrict__ mask,
    const int* __restrict__ pe, __bf16* __restrict__ ao) {
  const int lane = threadIdx.x & 31;
  const int wid = blockIdx.x * 8 + (threadIdx.x >> 5);
  const int i = wid % 5;
  const int m = (wid / 5) & 7;
  const int hw = (wid / 40) & 1023;
  const int b = wid / 40960;
  const int ti = pe[(b * 5 + i) * 3072 + 2];
  const float q = (float)Qb[((size_t)(b * 5 + i) * 1024 + hw) * 256 + m * 32 + lane];
  const size_t base = (size_t)(((b * 2 + ti) * 8 + m) * 5) * 32768 +
                      (size_t)hw * 32 + lane;
  float att[5];
#pragma unroll
  for (int j = 0; j < 5; ++j) {
    float s = q * (float)kt[base + (size_t)j * 32768];
#pragma unroll
    for (int o = 16; o > 0; o >>= 1) s += __shfl_xor(s, o, 32);
    s *= 0.17677669529663687f;  // 32^-0.5
    const int mk = mask[((size_t)(b * 1024 + hw) * 5 + i) * 5 + j];
    att[j] = (mk == 0) ? -1e9f : s;
  }
  float mx = att[0];
#pragma unroll
  for (int j = 1; j < 5; ++j) mx = fmaxf(mx, att[j]);
  float sum = 0.f;
#pragma unroll
  for (int j = 0; j < 5; ++j) { att[j] = __expf(att[j] - mx); sum += att[j]; }
  const float inv = 1.0f / sum;
  float acc = 0.f;
#pragma unroll
  for (int j = 0; j < 5; ++j)
    acc += att[j] * inv * (float)vt[base + (size_t)j * 32768];
  ao[((size_t)(b * 5 + i) * 1024 + hw) * 256 + m * 32 + lane] = (__bf16)acc;
}

// ---------------- final a-projection GEMM (f32 output) ------------------------

__global__ void __launch_bounds__(256) k_out(
    const __bf16* __restrict__ ao, const __bf16* __restrict__ wa,
    const float* __restrict__ ab, const int* __restrict__ pe,
    float* __restrict__ out) {
  const int lane = threadIdx.x & 31;
  const int tile = blockIdx.x * 8 + (threadIdx.x >> 5);
  const int mt = tile >> 4, nt = tile & 15;
  const int bl = blockIdx.y;
  const int t = pe[bl * 3072 + 2];
  const int arow = mt * 16 + (lane & 15);
  const int ncol = nt * 16 + (lane & 15);
  const __bf16* ap = ao + ((size_t)bl * 1024 + arow) * 256 + ((lane < 16) ? 0 : 16);
  const __bf16* bp = wa + (size_t)t * 65536 + (size_t)ncol * 256 + ((lane < 16) ? 0 : 16);
  f32x8 c = {};
#pragma unroll
  for (int k0 = 0; k0 < 256; k0 += 32) {
    bf16x16 a = load_frag_gap16(ap + k0);
    c = wmma_bf16(a, load_frag_contig(bp + k0), c);
  }
  const float bias = ab[t * 256 + ncol];
#pragma unroll
  for (int r = 0; r < 8; ++r) {
    const int mrow = mt * 16 + ((lane < 16) ? r : r + 8);
    out[((size_t)bl * 1024 + mrow) * 256 + ncol] = c[r] + bias;
  }
}

// ---------------- host ---------------------------------------------------------

extern "C" void kernel_launch(void* const* d_in, const int* in_sizes, int n_in,
                              void* d_out, int out_size, void* d_ws, size_t ws_size,
                              hipStream_t stream) {
  (void)in_sizes; (void)n_in; (void)out_size; (void)ws_size;
  const float* x    = (const float*)d_in[0];
  const int*   mask = (const int*)  d_in[1];
  const int*   pe   = (const int*)  d_in[2];
  const float* qw   = (const float*)d_in[3];
  const float* qb   = (const float*)d_in[4];
  const float* kw   = (const float*)d_in[5];
  const float* kb   = (const float*)d_in[6];
  const float* vw   = (const float*)d_in[7];
  const float* vb   = (const float*)d_in[8];
  const float* aw   = (const float*)d_in[9];
  const float* ab   = (const float*)d_in[10];
  const float* ra   = (const float*)d_in[11];
  const float* rm   = (const float*)d_in[12];
  const float* lw   = (const float*)d_in[13];
  const float* lb   = (const float*)d_in[14];
  float* out = (float*)d_out;

  char* w = (char*)d_ws;
  __bf16* xn  = (__bf16*)(w + 0);            // 10,485,760 B
  __bf16* Qb  = (__bf16*)(w + 10485760);     // 10,485,760
  __bf16* Kb  = (__bf16*)(w + 20971520);     // 10,485,760
  __bf16* Vb  = (__bf16*)(w + 31457280);     // 10,485,760
  __bf16* kt  = (__bf16*)(w + 41943040);     // 20,971,520
  __bf16* vt  = (__bf16*)(w + 62914560);     // 20,971,520
  __bf16* ao  = (__bf16*)(w + 83886080);     // 10,485,760
  __bf16* bwq = (__bf16*)(w + 94371840);     //    262,144
  __bf16* bwk = (__bf16*)(w + 94633984);
  __bf16* bwv = (__bf16*)(w + 94896128);
  __bf16* bwa = (__bf16*)(w + 95158272);
  __bf16* bra = (__bf16*)(w + 95420416);     //     65,536
  __bf16* brm = (__bf16*)(w + 95485952);     //     65,536  (transposed msg)

  k_cvt<<<512, 256, 0, stream>>>(qw, bwq, 131072);
  k_cvt<<<512, 256, 0, stream>>>(kw, bwk, 131072);
  k_cvt<<<512, 256, 0, stream>>>(vw, bwv, 131072);
  k_cvt<<<512, 256, 0, stream>>>(aw, bwa, 131072);
  k_cvt<<<128, 256, 0, stream>>>(ra, bra, 32768);
  k_trel<<<32, 1024, 0, stream>>>(rm, brm);
  k_ln<<<2560, 256, 0, stream>>>(x, lw, lb, xn);
  k_qkv<<<dim3(128, 20), 256, 0, stream>>>(xn, bwq, bwk, bwv, qb, kb, vb, pe,
                                           Qb, Kb, Vb);
  k_rel<<<dim3(16, 320, 2), 256, 0, stream>>>(Kb, Vb, bra, brm, pe, kt, vt);
  k_attn<<<20480, 256, 0, stream>>>(Qb, kt, vt, mask, pe, ao);
  k_out<<<dim3(128, 20), 256, 0, stream>>>(ao, bwa, ab, pe, out);
}
